// DoubleScaleAlignment_47863115546787
// MI455X (gfx1250) — compile-verified, matrix operand order checked by reading, not run on hardware
//
#include <hip/hip_runtime.h>
#include <hip/hip_bf16.h>

// ---------------------------------------------------------------------------
// DoubleScaleAlignment on gfx1250 (MI455X).
// Dense contractions run through a WMMA GEMM with:
//   A = activations, pixel-major [Npix][K]  -> contiguous b128 fragment loads
//   B = weights, pre-packed in exact CDNA5 B-fragment order -> one 32B load
// Each wave: one 16-pixel tile x all channel tiles (A-fragment reuse).
// ---------------------------------------------------------------------------

typedef __attribute__((ext_vector_type(16))) _Float16 v16h;
typedef __attribute__((ext_vector_type(8)))  _Float16 v8h;
typedef __attribute__((ext_vector_type(8)))  float    v8f;

#define CCH  64
#define HH   128
#define WW   128
#define H2   64
#define W2   64
#define OFFC 144            // G*2*KK
#define PIX  (HH*WW)        // 16384
#define PIX2 (H2*W2)        // 4096

// ------------------------------ elementwise --------------------------------

__global__ void copy_f32(const float* __restrict__ in, float* __restrict__ out, int n) {
    int i = blockIdx.x * blockDim.x + threadIdx.x;
    if (i < n) out[i] = in[i];
}

// x frame [64][128][128] -> pooled [64][64][64]
__global__ void avgpool2(const float* __restrict__ in, float* __restrict__ out) {
    int i = blockIdx.x * blockDim.x + threadIdx.x;
    if (i >= CCH * PIX2) return;
    int c = i >> 12, p = i & 4095;
    int y = p >> 6, x = p & 63;
    const float* q = in + (size_t)c * PIX + (y * 2) * WW + x * 2;
    out[i] = 0.25f * (q[0] + q[1] + q[WW] + q[WW + 1]);
}

// im2col 3x3 pad1 on 64x64, virtual concat [fn2, fc2] -> PIXEL-MAJOR [4096][1152]
__global__ void im2col3(const float* __restrict__ fn2, const float* __restrict__ fc2,
                        _Float16* __restrict__ col) {
    int i = blockIdx.x * blockDim.x + threadIdx.x;
    if (i >= 1152 * PIX2) return;
    int p = i / 1152, r = i % 1152;          // r minor -> coalesced writes
    int c = r / 9, k = r % 9;
    int y = (p >> 6) + k / 3 - 1;
    int x = (p & 63) + k % 3 - 1;
    float v = 0.f;
    if (y >= 0 && y < H2 && x >= 0 && x < W2) {
        const float* src = (c < CCH) ? (fn2 + (size_t)c * PIX2)
                                     : (fc2 + (size_t)(c - CCH) * PIX2);
        v = src[y * W2 + x];
    }
    col[i] = (_Float16)v;
}

// im2col 5x5 pad2 on 128x128, virtual concat [z1, fc1] -> PIXEL-MAJOR [16384][3200]
__global__ void im2col5(const float* __restrict__ z1, const float* __restrict__ fc1,
                        _Float16* __restrict__ col) {
    int i = blockIdx.x * blockDim.x + threadIdx.x;
    if (i >= 3200 * PIX) return;
    int p = i / 3200, r = i % 3200;
    int c = r / 25, k = r % 25;
    int y = (p >> 7) + k / 5 - 2;
    int x = (p & 127) + k % 5 - 2;
    float v = 0.f;
    if (y >= 0 && y < HH && x >= 0 && x < WW) {
        const float* src = (c < CCH) ? (z1 + (size_t)c * PIX)
                                     : (fc1 + (size_t)(c - CCH) * PIX);
        v = src[y * WW + x];
    }
    col[i] = (_Float16)v;
}

// bilinear 2x upsample (half-pixel centers, edge clamp) * 2.0
// in [144][64][64] -> out [144][128][128] (channel-major)
__global__ void upsample2x(const float* __restrict__ in, float* __restrict__ out) {
    int i = blockIdx.x * blockDim.x + threadIdx.x;
    if (i >= OFFC * PIX) return;
    int c = i >> 14, p = i & 16383;
    int y = p >> 7, x = p & 127;
    float syf = 0.5f * (float)y - 0.25f;
    float sxf = 0.5f * (float)x - 0.25f;
    int y0 = (int)floorf(syf); float wy = syf - (float)y0;
    int x0 = (int)floorf(sxf); float wx = sxf - (float)x0;
    int y0c = min(max(y0, 0), H2 - 1), y1c = min(max(y0 + 1, 0), H2 - 1);
    int x0c = min(max(x0, 0), W2 - 1), x1c = min(max(x0 + 1, 0), W2 - 1);
    const float* ch = in + (size_t)c * PIX2;
    float v = ch[y0c * W2 + x0c] * (1.f - wy) * (1.f - wx)
            + ch[y0c * W2 + x1c] * (1.f - wy) * wx
            + ch[y1c * W2 + x0c] * wy * (1.f - wx)
            + ch[y1c * W2 + x1c] * wy * wx;
    out[i] = 2.0f * v;
}

// deformable bilinear sampling -> PIXEL-MAJOR samp [16384][576]
// fn1 [64][128][128], off [144][128][128] (layout (g,k,{dy,dx}))
__device__ __forceinline__ float gather0(const float* ch, int yi, int xi) {
    if (yi < 0 || yi >= HH || xi < 0 || xi >= WW) return 0.f;
    return ch[yi * WW + xi];
}

__global__ void deform_sample(const float* __restrict__ fn1,
                              const float* __restrict__ off,
                              _Float16* __restrict__ samp) {
    int i = blockIdx.x * blockDim.x + threadIdx.x;
    if (i >= 576 * PIX) return;
    int p = i / 576, r = i % 576;            // r = g*72 + c*9 + k
    int k = r % 9;
    int c = (r / 9) & 7;
    int g = r / 72;
    int py = p >> 7, px = p & 127;
    float dy = off[(size_t)((g * 9 + k) * 2 + 0) * PIX + p];
    float dx = off[(size_t)((g * 9 + k) * 2 + 1) * PIX + p];
    float sy = dy + (float)py + (float)(k / 3 - 1);
    float sx = dx + (float)px + (float)(k % 3 - 1);
    float y0f = floorf(sy), x0f = floorf(sx);
    float wy = sy - y0f, wx = sx - x0f;
    int y0 = (int)y0f, x0 = (int)x0f;
    const float* ch = fn1 + (size_t)(g * 8 + c) * PIX;
    float v = gather0(ch, y0,     x0    ) * (1.f - wy) * (1.f - wx)
            + gather0(ch, y0,     x0 + 1) * (1.f - wy) * wx
            + gather0(ch, y0 + 1, x0    ) * wy * (1.f - wx)
            + gather0(ch, y0 + 1, x0 + 1) * wy * wx;
    samp[i] = (_Float16)v;
}

// ------------------------- weight fragment packing -------------------------
// w [Mch][K] row-major f32 -> pk[kb][tn][lane][16] f16, exact CDNA5 B-fragment
// order: lane n = l&15, koff8 = (l>>4)*8; elem j<8 -> k=kb*32+koff8+j,
//                                         elem j>=8 -> k=kb*32+16+koff8+(j-8)
__global__ void pack_Bfrag(const float* __restrict__ w, _Float16* __restrict__ pk,
                           int Mch, int K) {
    int i = blockIdx.x * blockDim.x + threadIdx.x;
    int tilesN = Mch >> 4;
    int total = (K >> 5) * tilesN * 512;
    if (i >= total) return;
    int e  = i & 15;
    int l  = (i >> 4) & 31;
    int tn = (i >> 9) % tilesN;
    int kb = i / (tilesN << 9);
    int n     = tn * 16 + (l & 15);
    int koff8 = (l >> 4) * 8;
    int k = kb * 32 + ((e >> 3) << 4) + koff8 + (e & 7);
    pk[i] = (_Float16)w[(size_t)n * K + k];
}

// ------------------------------- WMMA GEMM ---------------------------------
// outC[ch][ldc] = sum_k act[pix][k] * W[ch][k]  (+bias[ch]) (+extra[ch][ldc])
// act pixel-major [Npix][K]; pk packed B fragments; one wave = 16 pixels x
// TILES_N channel tiles; A fragment reused across channel tiles.
template <int TILES_N>
__global__ __launch_bounds__(128)
void wmma_gemm_pm(const _Float16* __restrict__ act,
                  const _Float16* __restrict__ pk,
                  float* __restrict__ outC,
                  const float* __restrict__ bias,
                  const float* __restrict__ extra,
                  int Npix, int K, int ldc) {
    const int lane = threadIdx.x & 31;
    const int wave = threadIdx.x >> 5;
    const int tileP = blockIdx.x * 4 + wave;        // wave-uniform
    if (tileP * 16 >= Npix) return;                 // uniform across wave

    const int half  = lane >> 4;
    const int r     = lane & 15;
    const int koff8 = half * 8;

    const _Float16* Arow = act + ((size_t)tileP * 16 + r) * K + koff8;
    const _Float16* Pkl  = pk + (size_t)lane * 16;

    v8f acc[TILES_N] = {};
    const int KB = K >> 5;
    for (int kb = 0; kb < KB; ++kb) {
        v8h alo = *reinterpret_cast<const v8h*>(Arow + kb * 32);
        v8h ahi = *reinterpret_cast<const v8h*>(Arow + kb * 32 + 16);
        v16h a = __builtin_shufflevector(alo, ahi,
                 0,1,2,3,4,5,6,7,8,9,10,11,12,13,14,15);
        const _Float16* pb = Pkl + (size_t)kb * TILES_N * 512;
#pragma unroll
        for (int tn = 0; tn < TILES_N; ++tn) {
            v16h b = *reinterpret_cast<const v16h*>(pb + (size_t)tn * 512);
            acc[tn] = __builtin_amdgcn_wmma_f32_16x16x32_f16(
                          false, a, false, b, (short)0, acc[tn], false, false);
        }
    }

    const int pixbase = tileP * 16 + half * 8;      // pixel m = v + 8*half
#pragma unroll
    for (int tn = 0; tn < TILES_N; ++tn) {
        int ch = tn * 16 + r;
        float bs = bias ? bias[ch] : 0.f;
        size_t rowoff = (size_t)ch * ldc + pixbase;
        if (extra) {
#pragma unroll
            for (int v = 0; v < 8; ++v)
                outC[rowoff + v] = acc[tn][v] + bs + extra[rowoff + v];
        } else {
#pragma unroll
            for (int v = 0; v < 8; ++v)
                outC[rowoff + v] = acc[tn][v] + bs;
        }
    }
}

// ------------------------------ orchestration ------------------------------

static inline int eb(int n) { return (n + 255) / 256; }

extern "C" void kernel_launch(void* const* d_in, const int* in_sizes, int n_in,
                              void* d_out, int out_size, void* d_ws, size_t ws_size,
                              hipStream_t stream) {
    (void)in_sizes; (void)n_in; (void)out_size; (void)ws_size;
    const float* x        = (const float*)d_in[0]; // [2][5][64][128][128]
    const float* w_off1   = (const float*)d_in[1]; // [144][128][5][5] -> [144][3200]
    const float* b_off1   = (const float*)d_in[2]; // [144]
    const float* w_off2   = (const float*)d_in[3]; // [144][128][3][3] -> [144][1152]
    const float* b_off2   = (const float*)d_in[4]; // [144]
    const float* w_align1 = (const float*)d_in[5]; // [64][64][3][3]   -> [64][576]
    const float* w_alignf = (const float*)d_in[6];
    float* out = (float*)d_out;

    size_t o = 0;
    auto carve = [&](size_t bytes) -> char* {
        char* p = (char*)d_ws + o;
        o += (bytes + 255) & ~(size_t)255;
        return p;
    };
    _Float16* pk1   = (_Float16*)carve((size_t)OFFC * 3200 * 2); // packed 5x5 W
    _Float16* pk2   = (_Float16*)carve((size_t)OFFC * 1152 * 2); // packed 3x3 W
    _Float16* pka1  = (_Float16*)carve((size_t)CCH * 576 * 2);   // packed align1
    _Float16* pkaf  = (_Float16*)carve((size_t)CCH * 576 * 2);   // packed alignf
    float*    fc2   = (float*)carve((size_t)CCH * PIX2 * 4);
    float*    fn2   = (float*)carve((size_t)CCH * PIX2 * 4);
    _Float16* im2   = (_Float16*)carve((size_t)PIX2 * 1152 * 2); // pixel-major
    float*    off2  = (float*)carve((size_t)OFFC * PIX2 * 4);
    float*    off2us= (float*)carve((size_t)OFFC * PIX * 4);
    _Float16* samp  = (_Float16*)carve((size_t)PIX * 576 * 2);   // pixel-major
    float*    z1    = (float*)carve((size_t)CCH * PIX * 4);
    _Float16* im5   = (_Float16*)carve((size_t)PIX * 3200 * 2);  // pixel-major
    float*    off1  = (float*)carve((size_t)OFFC * PIX * 4);

    // weight packs (flatten of conv weights is identity in memory order)
    pack_Bfrag<<<eb(OFFC * 3200), 256, 0, stream>>>(w_off1,   pk1,  OFFC, 3200);
    pack_Bfrag<<<eb(OFFC * 1152), 256, 0, stream>>>(w_off2,   pk2,  OFFC, 1152);
    pack_Bfrag<<<eb(CCH * 576),   256, 0, stream>>>(w_align1, pka1, CCH,  576);
    pack_Bfrag<<<eb(CCH * 576),   256, 0, stream>>>(w_alignf, pkaf, CCH,  576);

    const size_t frameSz = (size_t)CCH * PIX;
    for (int b = 0; b < 2; ++b) {
        const float* fc1 = x + ((size_t)b * 5 + 2) * frameSz;
        copy_f32<<<eb((int)frameSz), 256, 0, stream>>>(
            fc1, out + ((size_t)b * 5 + 2) * frameSz, (int)frameSz);
        avgpool2<<<eb(CCH * PIX2), 256, 0, stream>>>(fc1, fc2);

        for (int t = 0; t < 5; ++t) {
            if (t == 2) continue;
            const float* fn1 = x + ((size_t)b * 5 + t) * frameSz;

            // off2 = conv3x3(cat[fn2, fc2]) + bias           (M=144,K=1152,N=4096)
            avgpool2<<<eb(CCH * PIX2), 256, 0, stream>>>(fn1, fn2);
            im2col3<<<eb(1152 * PIX2), 256, 0, stream>>>(fn2, fc2, im2);
            wmma_gemm_pm<9><<<PIX2 / 64, 128, 0, stream>>>(
                im2, pk2, off2, b_off2, nullptr, PIX2, 1152, PIX2);

            // off2_us = bilinear_up2(off2) * 2
            upsample2x<<<eb(OFFC * PIX), 256, 0, stream>>>(off2, off2us);

            // z1 = deform_conv(fn1, off2_us, w_align1)       (M=64,K=576,N=16384)
            deform_sample<<<eb(576 * PIX), 256, 0, stream>>>(fn1, off2us, samp);
            wmma_gemm_pm<4><<<PIX / 64, 128, 0, stream>>>(
                samp, pka1, z1, nullptr, nullptr, PIX, 576, PIX);

            // off1 = conv5x5(cat[z1, fc1]) + bias + off2_us  (M=144,K=3200,N=16384)
            im2col5<<<eb(3200 * PIX), 256, 0, stream>>>(z1, fc1, im5);
            wmma_gemm_pm<9><<<PIX / 64, 128, 0, stream>>>(
                im5, pk1, off1, b_off1, off2us, PIX, 3200, PIX);

            // out = deform_conv(fn1, off1, w_alignf)
            deform_sample<<<eb(576 * PIX), 256, 0, stream>>>(fn1, off1, samp);
            wmma_gemm_pm<4><<<PIX / 64, 128, 0, stream>>>(
                samp, pkaf, out + ((size_t)b * 5 + t) * frameSz,
                nullptr, nullptr, PIX, 576, PIX);
        }
    }
}
